// CalorimeterCNN_62500364091854
// MI455X (gfx1250) — matrix operand order repaired and checked by reading, not compiled, not run on hardware
//
#include <hip/hip_runtime.h>
#include <hip/hip_bf16.h>
#include <math.h>

typedef __attribute__((ext_vector_type(16))) __bf16 bf16x16;
typedef __attribute__((ext_vector_type(8)))  __bf16 bf16x8;
typedef __attribute__((ext_vector_type(8)))  float  f32x8;

#define EPS 1e-5f
#define NB 512

__device__ __forceinline__ float bn_relu(float x, float g, float b, float m, float v) {
    float s = g * rsqrtf(v + EPS);
    float y = x * s + (b - m * s);
    return y > 0.f ? y : 0.f;
}

__device__ __forceinline__ bf16x16 make_frag(bf16x8 lo, bf16x8 hi) {
    return __builtin_shufflevector(lo, hi, 0, 1, 2, 3, 4, 5, 6, 7,
                                   8, 9, 10, 11, 12, 13, 14, 15);
}

// ---------------------------------------------------------------------------
// fp32 -> bf16 flat convert (weights)
// ---------------------------------------------------------------------------
__global__ void k_f32_to_bf16(const float* __restrict__ in, __bf16* __restrict__ out, int n) {
    int i = blockIdx.x * 256 + threadIdx.x;
    if (i < n) out[i] = (__bf16)in[i];
}

// ---------------------------------------------------------------------------
// Block 1: direct conv5x5(3->32) pad2 + BN + ReLU + maxpool2 -> bf16 [512][32][32][32]
// Cin=3 / K=75 is WMMA-hostile; VALU direct conv with LDS-cached weights.
// ---------------------------------------------------------------------------
__global__ __launch_bounds__(256)
void k_conv1_bn_pool(const float* __restrict__ x, const float* __restrict__ w,
                     const float* __restrict__ g, const float* __restrict__ bb,
                     const float* __restrict__ mn, const float* __restrict__ vr,
                     __bf16* __restrict__ out) {
    __shared__ float wsh[75];
    const int bi    = blockIdx.x;          // 512*32*4 blocks
    const int chunk = bi & 3;
    const int co    = (bi >> 2) & 31;
    const int b     = bi >> 7;
    if (threadIdx.x < 75) wsh[threadIdx.x] = w[co * 75 + threadIdx.x];
    __syncthreads();

    const int pp = chunk * 256 + threadIdx.x;   // pooled pixel 0..1023
    const int py = pp >> 5, px = pp & 31;
    float best = -1e30f;
    #pragma unroll
    for (int dy = 0; dy < 2; ++dy)
    #pragma unroll
    for (int dx = 0; dx < 2; ++dx) {
        const int oy = 2 * py + dy, ox = 2 * px + dx;
        float acc = 0.f;
        for (int ci = 0; ci < 3; ++ci) {
            const float* xp = x + ((b * 3 + ci) * 64) * 64;
            const float* wp = wsh + ci * 25;
            #pragma unroll
            for (int ky = 0; ky < 5; ++ky) {
                const int iy = oy + ky - 2;
                if (iy < 0 || iy >= 64) continue;
                #pragma unroll
                for (int kx = 0; kx < 5; ++kx) {
                    const int ix = ox + kx - 2;
                    if (ix < 0 || ix >= 64) continue;
                    acc += xp[iy * 64 + ix] * wp[ky * 5 + kx];
                }
            }
        }
        best = fmaxf(best, acc);
    }
    const float yv = bn_relu(best, g[co], bb[co], mn[co], vr[co]);
    out[((b * 32 + co) * 32 + py) * 32 + px] = (__bf16)yv;
}

// ---------------------------------------------------------------------------
// Implicit-GEMM conv3x3 (pad 1) + BN + ReLU + maxpool2 using WMMA bf16.
// GEMM: M = Cout (64), N = 64 image pixels (whole rows), K = CIN*9.
// 128 threads = 4 waves; wave w owns M rows [16w,16w+16), 4 WMMA n-tiles.
// Double-buffered LDS tiles (one barrier per K-step, loads overlap WMMA);
// [M][K]/[N][K] layouts + 40-elem row pad -> every fragment is 2x b128.
// ---------------------------------------------------------------------------
template<int CIN, int COUT, int HW>
__global__ __launch_bounds__(128)
void k_conv3x3_bn_pool_wmma(const __bf16* __restrict__ in,   // [B][CIN][HW][HW]
                            const __bf16* __restrict__ wk,   // [COUT][CIN*9]
                            const float* __restrict__ g, const float* __restrict__ bb,
                            const float* __restrict__ mn, const float* __restrict__ vr,
                            __bf16* __restrict__ out) {      // [B][COUT][HW/2][HW/2]
    constexpr int KT    = CIN * 9;
    constexpr int STEPS = KT / 32;
    constexpr int HWO   = HW / 2;

    __shared__ __bf16 At[2][64][40];   // ping-pong [M][K] + pad
    __shared__ __bf16 Bt[2][64][40];   // ping-pong [N][K] + pad
    __shared__ float  Ot[64][66];      // full-res conv tile + pad

    const int tid  = threadIdx.x;
    const int lane = tid & 31;
    const int wv   = tid >> 5;
    const int b    = blockIdx.z;
    const int co0  = blockIdx.y * 64;
    const int p0   = blockIdx.x * 64;      // 64 consecutive pixels = whole rows
    const int y0   = p0 / HW;

    // tile-fill coordinates (contiguous 16-element runs per thread)
    const int ar   = tid >> 1;             // A row, 0..63
    const int ac0  = (tid & 1) * 16;       // A col base
    const int bpx  = tid & 63;             // B row (pixel), 0..63
    const int bh   = tid >> 6;             // B K-half, 0..1
    const int bp   = p0 + bpx;
    const int by   = bp / HW, bxc = bp - by * HW;

    // fragment selectors
    const int halfA = (lane < 16) ? 0 : 1;
    const int ncol  = lane & 15;
    const int mrow  = 16 * wv + ncol;

    auto load_tile = [&](int st, int buf) {
        const int k0 = st * 32;
        #pragma unroll
        for (int i = 0; i < 16; ++i)
            At[buf][ar][ac0 + i] = wk[(co0 + ar) * KT + k0 + ac0 + i];
        #pragma unroll
        for (int i = 0; i < 16; ++i) {
            const int kg = k0 + bh * 16 + i;
            const int ci = kg / 9;
            const int rr = kg - ci * 9;
            const int ky = rr / 3 - 1;
            const int kx = rr - (rr / 3) * 3 - 1;
            const int iy = by + ky, ix = bxc + kx;
            __bf16 val = (__bf16)0.f;
            if (iy >= 0 && iy < HW && ix >= 0 && ix < HW)
                val = in[((b * CIN + ci) * HW + iy) * HW + ix];
            Bt[buf][bpx][bh * 16 + i] = val;
        }
    };

    f32x8 acc[4] = {};

    load_tile(0, 0);
    #pragma unroll 1
    for (int st = 0; st < STEPS; ++st) {
        __syncthreads();                   // tile 'st' visible; buf^1 free
        const int buf = st & 1;
        if (st + 1 < STEPS) load_tile(st + 1, buf ^ 1);

        // gather ALL fragments first, then fire WMMAs back-to-back
        const bf16x8* ap = (const bf16x8*)(&At[buf][mrow][0]);
        const bf16x16 a  = make_frag(ap[halfA], ap[2 + halfA]);
        bf16x16 bfr[4];
        #pragma unroll
        for (int t = 0; t < 4; ++t) {
            const bf16x8* bpnt = (const bf16x8*)(&Bt[buf][16 * t + ncol][0]);
            bfr[t] = make_frag(bpnt[halfA], bpnt[2 + halfA]);
        }
        #pragma unroll
        for (int t = 0; t < 4; ++t)
            acc[t] = __builtin_amdgcn_wmma_f32_16x16x32_bf16(false, a, false, bfr[t],
                                                             (short)0, acc[t], false, false);
    }
    __syncthreads();

    // accumulators -> LDS out tile (C/D layout: lane<16 -> M=v, lane>=16 -> M=v+8)
    #pragma unroll
    for (int t = 0; t < 4; ++t)
        #pragma unroll
        for (int vv = 0; vv < 8; ++vv) {
            const int rm = 16 * wv + ((lane < 16) ? vv : vv + 8);
            Ot[rm][16 * t + ncol] = acc[t][vv];
        }
    __syncthreads();

    // pool 2x2 (max before BN is valid: BN scale > 0, ReLU monotone) + BN + ReLU
    #pragma unroll
    for (int i = 0; i < 8; ++i) {
        const int q  = tid * 8 + i;               // 64 co x 16 pooled px
        const int co = q >> 4, pp = q & 15;
        const int py_l = pp / (HW / 2), px_l = pp % (HW / 2);
        const int r0 = 2 * py_l, c0 = 2 * px_l;
        float m4 = fmaxf(fmaxf(Ot[co][r0 * HW + c0],       Ot[co][r0 * HW + c0 + 1]),
                         fmaxf(Ot[co][(r0 + 1) * HW + c0], Ot[co][(r0 + 1) * HW + c0 + 1]));
        const int cog = co0 + co;
        const float yv = bn_relu(m4, g[cog], bb[cog], mn[cog], vr[cog]);
        const int pyg = y0 / 2 + py_l;
        out[((b * COUT + cog) * HWO + pyg) * HWO + px_l] = (__bf16)yv;
    }
}

// ---------------------------------------------------------------------------
// FC: s = relu(bn1d(f @ fc_w^T + fc_b)),  f [512][8192] bf16, fc_w [256][8192] bf16.
// [N][K] B tile: no transpose needed; double-buffered; prefetch 2 steps ahead.
// ---------------------------------------------------------------------------
__global__ __launch_bounds__(128)
void k_fc_bn_relu_wmma(const __bf16* __restrict__ f, const __bf16* __restrict__ wkb,
                       const float* __restrict__ bias,
                       const float* __restrict__ g, const float* __restrict__ bb,
                       const float* __restrict__ mn, const float* __restrict__ vr,
                       float* __restrict__ sout) {
    __shared__ __bf16 At[2][64][40];
    __shared__ __bf16 Bt[2][64][40];

    const int tid  = threadIdx.x;
    const int lane = tid & 31;
    const int wv   = tid >> 5;
    const int b0   = blockIdx.y * 64;   // batch rows
    const int n0   = blockIdx.x * 64;   // out features

    const int ar  = tid >> 1;
    const int ac0 = (tid & 1) * 16;
    const int bn_ = tid & 63;
    const int bh  = tid >> 6;

    const int halfA = (lane < 16) ? 0 : 1;
    const int ncol  = lane & 15;
    const int mrow  = 16 * wv + ncol;

    auto load_tile = [&](int st, int buf) {
        const int k0 = st * 32;
        #pragma unroll
        for (int i = 0; i < 16; ++i)
            At[buf][ar][ac0 + i] = f[(b0 + ar) * 8192 + k0 + ac0 + i];
        #pragma unroll
        for (int i = 0; i < 16; ++i)
            Bt[buf][bn_][bh * 16 + i] = wkb[(n0 + bn_) * 8192 + k0 + bh * 16 + i];
    };

    f32x8 acc[4] = {};

    load_tile(0, 0);
    #pragma unroll 1
    for (int st = 0; st < 256; ++st) {
        __syncthreads();
        const int buf = st & 1;
        if (st + 1 < 256) load_tile(st + 1, buf ^ 1);
        if (st + 2 < 256) {    // prefetch weight stream 2 steps ahead -> global_prefetch_b8
            __builtin_prefetch(&wkb[(n0 + bn_) * 8192 + (st + 2) * 32 + bh * 16], 0, 3);
            __builtin_prefetch(&f[(b0 + ar) * 8192 + (st + 2) * 32 + ac0], 0, 3);
        }

        const bf16x8* ap = (const bf16x8*)(&At[buf][mrow][0]);
        const bf16x16 a  = make_frag(ap[halfA], ap[2 + halfA]);
        bf16x16 bfr[4];
        #pragma unroll
        for (int t = 0; t < 4; ++t) {
            const bf16x8* bpnt = (const bf16x8*)(&Bt[buf][16 * t + ncol][0]);
            bfr[t] = make_frag(bpnt[halfA], bpnt[2 + halfA]);
        }
        #pragma unroll
        for (int t = 0; t < 4; ++t)
            acc[t] = __builtin_amdgcn_wmma_f32_16x16x32_bf16(false, a, false, bfr[t],
                                                             (short)0, acc[t], false, false);
    }

    #pragma unroll
    for (int t = 0; t < 4; ++t)
        #pragma unroll
        for (int vv = 0; vv < 8; ++vv) {
            const int row = b0 + 16 * wv + ((lane < 16) ? vv : vv + 8);
            const int col = n0 + 16 * t + ncol;
            const float yv = bn_relu(acc[t][vv] + bias[col], g[col], bb[col], mn[col], vr[col]);
            sout[row * 256 + col] = yv;
        }
}

// ---------------------------------------------------------------------------
// Heads (tiny): hid[h][b][j] = relu(bn(s[b,:]·w1[j,:] + b1));  out = hid·w2 + b2
// ---------------------------------------------------------------------------
__global__ void k_head1(const float* __restrict__ s,
                        const float* w1c, const float* b1c,
                        const float* gc, const float* bbc, const float* mc, const float* vc,
                        const float* w1m, const float* b1m,
                        const float* gm, const float* bbm, const float* mm, const float* vm,
                        float* __restrict__ hid) {
    int i = blockIdx.x * 256 + threadIdx.x;        // 2*512*64
    if (i >= 2 * NB * 64) return;
    const int j = i & 63, b = (i >> 6) & (NB - 1), h = i >> 15;
    const float* w  = h ? w1m : w1c;
    const float* bi = h ? b1m : b1c;
    const float* gg = h ? gm : gc;   const float* bbv = h ? bbm : bbc;
    const float* mv = h ? mm : mc;   const float* vv  = h ? vm : vc;
    float acc = bi[j];
    const float* sr = s + b * 256;
    const float* wr = w + j * 256;
    for (int k = 0; k < 256; ++k) acc += sr[k] * wr[k];
    hid[i] = bn_relu(acc, gg[j], bbv[j], mv[j], vv[j]);
}

__global__ void k_head2(const float* __restrict__ hid,
                        const float* w2c, const float* b2c,
                        const float* w2m, const float* b2m,
                        float* __restrict__ out) {
    int i = blockIdx.x * 256 + threadIdx.x;        // 2*512
    if (i >= 2 * NB) return;
    const int b = i & (NB - 1), h = i >> 9;
    const float* w = h ? w2m : w2c;
    float acc = (h ? b2m : b2c)[0];
    const float* hr = hid + (size_t)(h * NB + b) * 64;
    #pragma unroll
    for (int k = 0; k < 64; ++k) acc += hr[k] * w[k];
    out[h * NB + b] = acc;                          // d_out = [c | mo]
}

// ---------------------------------------------------------------------------
extern "C" void kernel_launch(void* const* d_in, const int* in_sizes, int n_in,
                              void* d_out, int out_size, void* d_ws, size_t ws_size,
                              hipStream_t stream) {
    const float* x     = (const float*)d_in[0];
    const float* w1    = (const float*)d_in[1];
    const float *bn1g = (const float*)d_in[2], *bn1b = (const float*)d_in[3],
                *bn1m = (const float*)d_in[4], *bn1v = (const float*)d_in[5];
    const float* w2    = (const float*)d_in[6];
    const float *bn2g = (const float*)d_in[7], *bn2b = (const float*)d_in[8],
                *bn2m = (const float*)d_in[9], *bn2v = (const float*)d_in[10];
    const float* w3    = (const float*)d_in[11];
    const float *bn3g = (const float*)d_in[12], *bn3b = (const float*)d_in[13],
                *bn3m = (const float*)d_in[14], *bn3v = (const float*)d_in[15];
    const float* fcw   = (const float*)d_in[16];
    const float* fcb   = (const float*)d_in[17];
    const float *bnfg = (const float*)d_in[18], *bnfb = (const float*)d_in[19],
                *bnfm = (const float*)d_in[20], *bnfv = (const float*)d_in[21];
    const float* cw1   = (const float*)d_in[22];
    const float* cb1   = (const float*)d_in[23];
    const float *bncg = (const float*)d_in[24], *bncb = (const float*)d_in[25],
                *bncm = (const float*)d_in[26], *bncv = (const float*)d_in[27];
    const float* cw2   = (const float*)d_in[28];
    const float* cb2   = (const float*)d_in[29];
    const float* mw1   = (const float*)d_in[30];
    const float* mb1   = (const float*)d_in[31];
    const float *bnmg = (const float*)d_in[32], *bnmb = (const float*)d_in[33],
                *bnmm = (const float*)d_in[34], *bnmv = (const float*)d_in[35];
    const float* mw2   = (const float*)d_in[36];
    const float* mb2   = (const float*)d_in[37];
    float* out = (float*)d_out;

    // workspace layout (256B aligned)
    char* ws = (char*)d_ws;
    size_t off = 0;
    auto carve = [&](size_t bytes) -> void* {
        void* p = ws + off;
        off = (off + bytes + 255) & ~(size_t)255;
        return p;
    };
    __bf16* h1   = (__bf16*)carve((size_t)NB * 32 * 32 * 32 * 2);   // 33.5 MB
    __bf16* h2   = (__bf16*)carve((size_t)NB * 64 * 16 * 16 * 2);   // 16.8 MB
    __bf16* h3   = (__bf16*)carve((size_t)NB * 128 * 8 * 8 * 2);    //  8.4 MB
    __bf16* w2b  = (__bf16*)carve((size_t)64 * 288 * 2);
    __bf16* w3b  = (__bf16*)carve((size_t)128 * 576 * 2);
    __bf16* fcwb = (__bf16*)carve((size_t)256 * 8192 * 2);          //  4.2 MB
    float*  sbuf = (float*)carve((size_t)NB * 256 * 4);
    float*  hid  = (float*)carve((size_t)2 * NB * 64 * 4);

    // weight conversion (deterministic each call)
    k_f32_to_bf16<<<(64 * 288 + 255) / 256, 256, 0, stream>>>(w2, w2b, 64 * 288);
    k_f32_to_bf16<<<(128 * 576 + 255) / 256, 256, 0, stream>>>(w3, w3b, 128 * 576);
    k_f32_to_bf16<<<(256 * 8192 + 255) / 256, 256, 0, stream>>>(fcw, fcwb, 256 * 8192);

    // block 1: direct conv + bn + relu + pool
    k_conv1_bn_pool<<<NB * 32 * 4, 256, 0, stream>>>(x, w1, bn1g, bn1b, bn1m, bn1v, h1);

    // block 2: WMMA implicit GEMM conv3x3 32->64 @32x32, pool -> 16x16
    {
        dim3 grid(32 * 32 / 64, 64 / 64, NB);
        k_conv3x3_bn_pool_wmma<32, 64, 32><<<grid, 128, 0, stream>>>(
            h1, w2b, bn2g, bn2b, bn2m, bn2v, h2);
    }
    // block 3: WMMA implicit GEMM conv3x3 64->128 @16x16, pool -> 8x8
    {
        dim3 grid(16 * 16 / 64, 128 / 64, NB);
        k_conv3x3_bn_pool_wmma<64, 128, 16><<<grid, 128, 0, stream>>>(
            h2, w3b, bn3g, bn3b, bn3m, bn3v, h3);
    }
    // FC 8192 -> 256 (+bias, bn1d, relu)
    {
        dim3 grid(256 / 64, NB / 64);
        k_fc_bn_relu_wmma<<<grid, 128, 0, stream>>>(
            (const __bf16*)h3, fcwb, fcb, bnfg, bnfb, bnfm, bnfv, sbuf);
    }
    // heads
    k_head1<<<(2 * NB * 64 + 255) / 256, 256, 0, stream>>>(
        sbuf, cw1, cb1, bncg, bncb, bncm, bncv,
        mw1, mb1, bnmg, bnmb, bnmm, bnmv, hid);
    k_head2<<<(2 * NB + 255) / 256, 256, 0, stream>>>(hid, cw2, cb2, mw2, mb2, out);

    (void)in_sizes; (void)n_in; (void)out_size; (void)ws_size;
}
